// MultiCNet_31945966748154
// MI455X (gfx1250) — matrix operand order repaired and checked by reading, not compiled, not run on hardware
//
#include <hip/hip_runtime.h>
#include <hip/hip_bf16.h>
#include <math.h>

// ---------------------------------------------------------------------------
// MultiCNet SDE scan for gfx1250 (MI455X).
// B=8192, D=64, H=512, T=101. Each wave owns a 16-row WMMA M-tile of the
// batch for all 100 steps; state lives in registers in WMMA C layout.
// W1+W2 double-buffered in LDS via async global->LDS DMA, prefetched one
// MLP ahead so the ASYNCcnt wait overlaps a full MLP of WMMA work.
// ---------------------------------------------------------------------------

typedef __attribute__((ext_vector_type(16))) __bf16 v16bf;
typedef __attribute__((ext_vector_type(8)))  __bf16 v8bf;
typedef __attribute__((ext_vector_type(8)))  float  v8f;
typedef __attribute__((ext_vector_type(4)))  int    v4i;

#define BATCH   8192
#define DDIM    64
#define HDIM    512
#define NSTEP   100
#define ROWS_PER_BLOCK 128   // 8 waves * 16 rows
#define W1_ELEMS (HDIM * DDIM)   // 32768 bf16 = 64 KB per MLP
#define W2_ELEMS (DDIM * HDIM)   // 32768 bf16 = 64 KB per MLP

// ---- workspace layout (bytes) ----
#define WS_W1P   0                       // [3][512][64]  bf16 (W1 transposed, K=64 x-part)
#define WS_W2P   196608                  // [3][64][512]  bf16 (W2 transposed)
#define WS_W1T   393216                  // [3][512] f32  (t-row of W1)
#define WS_B1P   399360                  // [3][512] f32
#define WS_B2S   405504                  // [64] f32 (bi2+ba2)
#define WS_B2D   405760                  // [64] f32 (bd2)

// ---- LDS layout (bytes) ----
#define LDS_W1B   0                          // [2][512][64] bf16 = 131072 (ping-pong)
#define LDS_W2B   131072                     // [2][64][512] bf16 = 131072 (ping-pong)
#define LDS_X     (131072 + 131072)          // 8 waves * [16][64] bf16 = 16384
#define LDS_H     (131072 + 131072 + 16384)  // 8 waves * [16][32] bf16 = 8192
#define LDS_E     (131072 + 131072 + 16384 + 8192) // [128] f32 = 512
#define LDS_TOTAL (131072 + 131072 + 16384 + 8192 + 512)  // 287232 B < 320 KB

__device__ __forceinline__ v8f wmma_bf16(v16bf a, v16bf b, v8f c) {
    // D = A(16x32 bf16) * B(32x16 bf16) + C(16x16 f32)
    return __builtin_amdgcn_wmma_f32_16x16x32_bf16(
        /*neg_a=*/false, a, /*neg_b=*/false, b,
        /*c_mod=*/(short)0, c, /*reuse_a=*/false, /*reuse_b=*/false);
}

// A-matrix 16x32 bf16 fragment from a row-major [16][rowStride] tile.
// ISA layout: lanes 0-15 -> row=lane, K = kOff+{0..7, 16..23};
//             lanes 16-31 -> row=lane-16, K = kOff+{8..15, 24..31}.
__device__ __forceinline__ v16bf load_frag_a(const __bf16* base, int rowStride,
                                             int lane, int kOff) {
    int row = lane & 15;
    int k0  = kOff + ((lane & 16) ? 8 : 0);
    const __bf16* p = base + row * rowStride + k0;
    v16bf f;
    v8bf* h = (v8bf*)&f;
    h[0] = *(const v8bf*)(p);        // 16B -> ds_load_b128
    h[1] = *(const v8bf*)(p + 16);
    return f;
}

// B-matrix 32x16 bf16 fragment from an [N][kStride] (column-major weight) tile.
// ISA layout: lane -> column N = nOff + lane%16; lanes 0-15 hold K=kOff+{0..7, 16..23},
// lanes 16-31 hold K=kOff+{8..15, 24..31}, 2 values per VGPR ascending K.
__device__ __forceinline__ v16bf load_frag_b(const __bf16* base, int kStride,
                                             int lane, int nOff, int kOff) {
    int n  = nOff + (lane & 15);
    int k0 = kOff + ((lane & 16) ? 8 : 0);
    const __bf16* p = base + n * kStride + k0;
    v16bf f;
    v8bf* h = (v8bf*)&f;
    h[0] = *(const v8bf*)(p);
    h[1] = *(const v8bf*)(p + 16);
    return f;
}

__device__ __forceinline__ float fast_tanh(float x) {
#if __has_builtin(__builtin_amdgcn_tanhf)
    return __builtin_amdgcn_tanhf(x);   // gfx1250 v_tanh_f32 (confirmed round 1)
#else
    return tanhf(x);
#endif
}

// CDNA5 async global->LDS copy of one 16-byte chunk per lane
// (ASYNCcnt-tracked). Builtin signature validated in rounds 2/3.
__device__ __forceinline__ void async_copy_b128(const void* gsrc, void* ldst) {
#if __has_builtin(__builtin_amdgcn_global_load_async_to_lds_b128)
    __builtin_amdgcn_global_load_async_to_lds_b128(
        (__attribute__((address_space(1))) v4i*)gsrc,
        (__attribute__((address_space(3))) v4i*)ldst,
        /*offset=*/0, /*cpol=*/0);
#else
    *(v8bf*)ldst = *(const v8bf*)gsrc;   // synchronous fallback
#endif
}

// Wait until at most N async-LDS ops remain (in-order completion => the
// older DMA batch has fully landed while the prefetch batch stays in flight).
__device__ __forceinline__ void wait_async_32() {
#if __has_builtin(__builtin_amdgcn_global_load_async_to_lds_b128)
#if __has_builtin(__builtin_amdgcn_s_wait_asynccnt)
    __builtin_amdgcn_s_wait_asynccnt(32);
#else
    asm volatile("s_wait_asynccnt 32" ::: "memory");
#endif
#endif
}

// ---------------------------------------------------------------------------
// One-time weight preparation: fp32 -> bf16, transposed to [N][K] for direct
// B-fragment loads; fold biases.
// ---------------------------------------------------------------------------
__global__ void prep_weights(const float* __restrict__ Wi1, const float* __restrict__ bi1,
                             const float* __restrict__ Wi2, const float* __restrict__ bi2,
                             const float* __restrict__ Wa1, const float* __restrict__ ba1,
                             const float* __restrict__ Wa2, const float* __restrict__ ba2,
                             const float* __restrict__ Wd1, const float* __restrict__ bd1,
                             const float* __restrict__ Wd2, const float* __restrict__ bd2,
                             __bf16* __restrict__ w1p, float* __restrict__ w1t,
                             float* __restrict__ b1p, __bf16* __restrict__ w2p,
                             float* __restrict__ b2s, float* __restrict__ b2d) {
    int tid = blockIdx.x * blockDim.x + threadIdx.x;
    const float* W1s[3] = {Wi1, Wa1, Wd1};
    const float* b1s[3] = {bi1, ba1, bd1};
    const float* W2s[3] = {Wi2, Wa2, Wd2};

    if (tid < 3 * HDIM) {                    // one thread per (mlp, hidden unit)
        int m = tid >> 9, n = tid & 511;
        const float* W1 = W1s[m];
        #pragma unroll 4
        for (int k = 0; k < DDIM; ++k)
            w1p[(size_t)(m * HDIM + n) * DDIM + k] = (__bf16)W1[k * HDIM + n];
        w1t[m * HDIM + n] = W1[DDIM * HDIM + n];   // row 64 = t coefficient
        b1p[m * HDIM + n] = b1s[m][n];
    }
    if (tid < 3 * DDIM) {                    // one thread per (mlp, out unit)
        int m = tid >> 6, n = tid & 63;
        const float* W2 = W2s[m];
        #pragma unroll 4
        for (int k = 0; k < HDIM; ++k)
            w2p[(size_t)(m * DDIM + n) * HDIM + k] = (__bf16)W2[k * DDIM + n];
    }
    if (tid < DDIM) {
        b2s[tid] = bi2[tid] + ba2[tid];
        b2d[tid] = bd2[tid];
    }
}

// ---------------------------------------------------------------------------
// Main persistent scan kernel. grid = 64 blocks x 256 threads (8 waves).
// ---------------------------------------------------------------------------
__global__ __launch_bounds__(256, 1)
void sde_scan(const float* __restrict__ tarr, const float* __restrict__ x0,
              const float* __restrict__ noise,
              const __bf16* __restrict__ w1p, const float* __restrict__ w1t,
              const float* __restrict__ b1p, const __bf16* __restrict__ w2p,
              const float* __restrict__ b2s, const float* __restrict__ b2d,
              float* __restrict__ out) {
    extern __shared__ __align__(16) char smem[];
    __bf16* ldsW1 = (__bf16*)(smem + LDS_W1B);   // [2][W1_ELEMS]
    __bf16* ldsW2 = (__bf16*)(smem + LDS_W2B);   // [2][W2_ELEMS]
    __bf16* ldsX  = (__bf16*)(smem + LDS_X);
    __bf16* ldsH  = (__bf16*)(smem + LDS_H);
    float*  ldsE  = (float*) (smem + LDS_E);

    const int tid  = threadIdx.x;
    const int lane = tid & 31;
    const int wv   = tid >> 5;
    const int colL = lane & 15;          // N within a 16-wide tile (C layout)
    const int rsel = (lane >> 4) & 1;    // C layout: lanes 16-31 hold rows M+8
    const int rowBase = blockIdx.x * ROWS_PER_BLOCK + wv * 16;

    if (tid < ROWS_PER_BLOCK) ldsE[tid] = 0.0f;

    // ---- prologue: kick off DMA of MLP 0 weights into buffer 0 ----
    #pragma unroll
    for (int j = 0; j < 16; ++j) {
        int idx = (tid + j * 256) * 8;                     // element offset
        async_copy_b128(w1p + idx, ldsW1 + idx);
        async_copy_b128(w2p + idx, ldsW2 + idx);
    }

    __bf16* myX = ldsX + wv * 16 * 64;   // wave-private A staging for GEMM1
    __bf16* myH = ldsH + wv * 16 * 32;   // wave-private A staging for GEMM2

    // ---- load initial state into C-layout registers; emit traj[0] ----
    v8f xst[4];
    #pragma unroll
    for (int nt = 0; nt < 4; ++nt) {
        int col = nt * 16 + colL;
        #pragma unroll
        for (int r = 0; r < 8; ++r) {
            int row = rowBase + r + rsel * 8;
            float xv = x0[row * DDIM + col];
            xst[nt][r] = xv;
            out[(size_t)row * 65 + col] = xv;
        }
    }
    if (colL == 0) {
        #pragma unroll
        for (int r = 0; r < 8; ++r)
            out[(size_t)(rowBase + r + rsel * 8) * 65 + 64] = 0.0f;
    }

    // ---- time scan ----
    for (int step = 0; step < NSTEP; ++step) {
        float tn  = tarr[step];
        float dtv = tarr[step + 1] - tn;
        float sq  = sqrtf(dtv);

        // prefetch this step's noise lines (gfx1250 global_prefetch_b8)
        __builtin_prefetch(&noise[((size_t)step * BATCH + rowBase + (lane >> 1))
                                  * DDIM + (lane & 1) * 32], 0, 1);

        // write current x as bf16 A-tile (wave-private, row-major [16][64])
        #pragma unroll
        for (int nt = 0; nt < 4; ++nt) {
            int col = nt * 16 + colL;
            #pragma unroll
            for (int r = 0; r < 8; ++r)
                myX[(r + rsel * 8) * 64 + col] = (__bf16)xst[nt][r];
        }
        v16bf af0 = load_frag_a(myX, 64, lane, 0);
        v16bf af1 = load_frag_a(myX, 64, lane, 32);

        // output accumulators seeded with folded output biases
        v8f dr[4], gg[4];
        #pragma unroll
        for (int nt2 = 0; nt2 < 4; ++nt2) {
            float bv = b2s[nt2 * 16 + colL];
            float gv = b2d[nt2 * 16 + colL];
            #pragma unroll
            for (int i = 0; i < 8; ++i) { dr[nt2][i] = bv; gg[nt2][i] = gv; }
        }

        #pragma unroll 1
        for (int m = 0; m < 3; ++m) {
            const int g   = step * 3 + m;   // global MLP iteration
            const int cur = g & 1;          // buffer holding MLP g's weights
            const int nxt = cur ^ 1;
            const int pm  = (g + 1) % 3;    // next MLP (wraps to next step)

            // ping buffer's previous readers are done -> safe to overwrite
            __syncthreads();

            // prefetch next MLP's weights (32 async b128 ops per wave)
            {
                const __bf16* w1g = w1p + (size_t)pm * W1_ELEMS;
                const __bf16* w2g = w2p + (size_t)pm * W2_ELEMS;
                __bf16* w1l = ldsW1 + nxt * W1_ELEMS;
                __bf16* w2l = ldsW2 + nxt * W2_ELEMS;
                #pragma unroll
                for (int j = 0; j < 16; ++j) {
                    int idx = (tid + j * 256) * 8;
                    async_copy_b128(w1g + idx, w1l + idx);
                    async_copy_b128(w2g + idx, w2l + idx);
                }
            }

            // current batch (issued one MLP ago) has landed; prefetch stays
            // in flight behind this whole MLP's WMMA work.
            wait_async_32();
            __syncthreads();

            const __bf16* W1m = ldsW1 + cur * W1_ELEMS;
            const __bf16* W2m = ldsW2 + cur * W2_ELEMS;
            const float*  b1m = b1p + m * HDIM;
            const float*  wtm = w1t + m * HDIM;

            #pragma unroll 1
            for (int np = 0; np < 16; ++np) {   // 32 hidden cols per iter
                #pragma unroll
                for (int ntl = 0; ntl < 2; ++ntl) {
                    int n   = np * 2 + ntl;
                    int col = n * 16 + colL;
                    float bias = b1m[col] + tn * wtm[col];  // t-column folded
                    v8f h;
                    #pragma unroll
                    for (int i = 0; i < 8; ++i) h[i] = bias;
                    h = wmma_bf16(af0, load_frag_b(W1m, 64, lane, n * 16, 0),  h);
                    h = wmma_bf16(af1, load_frag_b(W1m, 64, lane, n * 16, 32), h);
                    // tanh + bf16 pack into GEMM2 A staging (row-major [16][32])
                    #pragma unroll
                    for (int r = 0; r < 8; ++r) {
                        float hv = fast_tanh(h[r]);
                        myH[(r + rsel * 8) * 32 + ntl * 16 + colL] = (__bf16)hv;
                    }
                }
                // consume the 32 fresh hidden cols as one GEMM2 k-tile
                v16bf a2 = load_frag_a(myH, 32, lane, 0);
                #pragma unroll
                for (int nt2 = 0; nt2 < 4; ++nt2) {
                    v16bf bfr = load_frag_b(W2m, HDIM, lane, nt2 * 16, np * 32);
                    if (m == 2) gg[nt2] = wmma_bf16(a2, bfr, gg[nt2]);
                    else        dr[nt2] = wmma_bf16(a2, bfr, dr[nt2]);
                }
            }
        }

        // ---- energy = row-sum(drift^2): per-lane partials + 16-lane xor tree
        float part[8];
        #pragma unroll
        for (int r = 0; r < 8; ++r) {
            float s = 0.0f;
            #pragma unroll
            for (int nt2 = 0; nt2 < 4; ++nt2) { float d = dr[nt2][r]; s += d * d; }
            s += __shfl_xor(s, 1);
            s += __shfl_xor(s, 2);
            s += __shfl_xor(s, 4);
            s += __shfl_xor(s, 8);   // halves stay separate: rows M and M+8
            part[r] = s;
        }

        // ---- Euler-Maruyama update + trajectory store ----
        size_t outRow = (size_t)(step + 1) * BATCH;
        size_t nzRow  = (size_t)step * BATCH;
        #pragma unroll
        for (int nt2 = 0; nt2 < 4; ++nt2) {
            int col = nt2 * 16 + colL;
            #pragma unroll
            for (int r = 0; r < 8; ++r) {
                int row = rowBase + r + rsel * 8;
                float z  = noise[(nzRow + row) * DDIM + col];
                float gv = 1.0f / (1.0f + __expf(-gg[nt2][r]));  // sigmoid
                float xn = xst[nt2][r] + dr[nt2][r] * dtv + gv * sq * z;
                xst[nt2][r] = xn;
                out[(outRow + row) * 65 + col] = xn;
            }
        }
        if (colL == 0) {   // lanes 0 and 16 own rows M and M+8
            #pragma unroll
            for (int r = 0; r < 8; ++r) {
                int lrow = wv * 16 + r + rsel * 8;
                float e = ldsE[lrow] + part[r] * dtv;
                ldsE[lrow] = e;
                out[(outRow + rowBase + r + rsel * 8) * 65 + 64] = e;
            }
        }
    }
}

// ---------------------------------------------------------------------------
extern "C" void kernel_launch(void* const* d_in, const int* in_sizes, int n_in,
                              void* d_out, int out_size, void* d_ws, size_t ws_size,
                              hipStream_t stream) {
    const float* t     = (const float*)d_in[0];
    const float* x0    = (const float*)d_in[1];
    const float* noise = (const float*)d_in[2];
    const float* Wi1 = (const float*)d_in[3];  const float* bi1 = (const float*)d_in[4];
    const float* Wi2 = (const float*)d_in[5];  const float* bi2 = (const float*)d_in[6];
    const float* Wa1 = (const float*)d_in[7];  const float* ba1 = (const float*)d_in[8];
    const float* Wa2 = (const float*)d_in[9];  const float* ba2 = (const float*)d_in[10];
    const float* Wd1 = (const float*)d_in[11]; const float* bd1 = (const float*)d_in[12];
    const float* Wd2 = (const float*)d_in[13]; const float* bd2 = (const float*)d_in[14];

    char* ws = (char*)d_ws;
    __bf16* w1p = (__bf16*)(ws + WS_W1P);
    __bf16* w2p = (__bf16*)(ws + WS_W2P);
    float*  w1t = (float*) (ws + WS_W1T);
    float*  b1p = (float*) (ws + WS_B1P);
    float*  b2s = (float*) (ws + WS_B2S);
    float*  b2d = (float*) (ws + WS_B2D);

    prep_weights<<<6, 256, 0, stream>>>(Wi1, bi1, Wi2, bi2, Wa1, ba1, Wa2, ba2,
                                        Wd1, bd1, Wd2, bd2,
                                        w1p, w1t, b1p, w2p, b2s, b2d);

    sde_scan<<<BATCH / ROWS_PER_BLOCK, 256, LDS_TOTAL, stream>>>(
        t, x0, noise, w1p, w1t, b1p, w2p, b2s, b2d, (float*)d_out);
}
